// MultiHeadAttention_45930380263713
// MI455X (gfx1250) — matrix operand order repaired and checked by reading, not compiled, not run on hardware
//
#include <hip/hip_runtime.h>
#include <hip/hip_bf16.h>
#include <math.h>

#define DM 1024
#define SQ 2048
#define NB 2
#define NH 16
#define DK 64

typedef __attribute__((ext_vector_type(16))) _Float16 v16h;
typedef __attribute__((ext_vector_type(8)))  float    v8f;
typedef __attribute__((ext_vector_type(4)))  _Float16 h4;

// ---------------------------------------------------------------------------
// WMMA helpers (CDNA5 V_WMMA_F32_16X16X32_F16, wave32)
// ---------------------------------------------------------------------------
__device__ __forceinline__ v8f wmma_f16(v16h a, v16h b, v8f c) {
  return __builtin_amdgcn_wmma_f32_16x16x32_f16(
      /*neg_a=*/false, a, /*neg_b=*/false, b,
      /*c_mod=*/(short)0, c, /*reuse_a=*/false, /*reuse_b=*/false);
}

// A fragment: 16x32 f16 tile, row-major in LDS, row stride `ld` (elements).
// ISA layout: lane L holds row M=L%16; VGPR v holds K = ((v<4)?0:16) +
// ((L<16)?0:8) + 2*(v%4) + {0,1}.
__device__ __forceinline__ v16h load_a_frag_f16(const _Float16* p, int ld) {
  const int lane = threadIdx.x & 31;
  const int m = lane & 15;
  const int kb = (lane & 16) ? 8 : 0;
  const _Float16* row = p + m * ld;
  v16h a;
#pragma unroll
  for (int v = 0; v < 8; ++v) {
    const int k = ((v & 4) ? 16 : 0) + kb + 2 * (v & 3);
    a[2 * v]     = row[k];
    a[2 * v + 1] = row[k + 1];
  }
  return a;
}

// Same fragment but source data is f32 in LDS (converted on the fly).
__device__ __forceinline__ v16h load_a_frag_f32(const float* p, int ld) {
  const int lane = threadIdx.x & 31;
  const int m = lane & 15;
  const int kb = (lane & 16) ? 8 : 0;
  const float* row = p + m * ld;
  v16h a;
#pragma unroll
  for (int v = 0; v < 8; ++v) {
    const int k = ((v & 4) ? 16 : 0) + kb + 2 * (v & 3);
    a[2 * v]     = (_Float16)row[k];
    a[2 * v + 1] = (_Float16)row[k + 1];
  }
  return a;
}

// B fragment: logical B is KxN (K=32, N=16); LDS holds B-transposed (NxK
// row-major, row stride `ld`) so the two halves of each VGPR are contiguous.
// ISA layout: lane L holds col N=L%16; VGPR v holds K = ((v<4)?0:16) + ((L<16)?0:8) + 2*(v%4) + {0,1}.
__device__ __forceinline__ v16h load_b_frag_bt(const _Float16* p, int ld) {
  const int lane = threadIdx.x & 31;
  const int n = lane & 15;
  const int kb = (lane & 16) ? 8 : 0;
  const _Float16* row = p + n * ld;
  v16h b;
#pragma unroll
  for (int v = 0; v < 8; ++v) {
    const int k = ((v & 4) ? 16 : 0) + kb + 2 * (v & 3);
    b[2 * v]     = row[k];
    b[2 * v + 1] = row[k + 1];
  }
  return b;
}

// Async copy 16 bytes global -> LDS (CDNA5 GLOBAL_LOAD_ASYNC_TO_LDS_B128).
// Tracked by ASYNCcnt; no VGPR round-trip.
__device__ __forceinline__ void async_copy_b128(const void* gptr, void* lptr) {
  const unsigned ldsOff = (unsigned)(unsigned long long)lptr;  // LDS addr[31:0]
  asm volatile("global_load_async_to_lds_b128 %0, %1, off"
               :
               : "v"(ldsOff), "v"(gptr)
               : "memory");
}

__device__ __forceinline__ void async_wait0() {
  asm volatile("s_wait_asynccnt 0x0" ::: "memory");
}

// ---------------------------------------------------------------------------
// GEMM: Out[4096 x 1024] = A[4096 x 1024](f32) @ W[1024 x 1024](f32) + bias.
// Block = 128 threads (4 waves), 64x64 macro-tile, K-chunk = 32 (one WMMA).
// Writes f16 (QKV path) if outH != nullptr, else f32 (final projection).
// ---------------------------------------------------------------------------
__global__ __launch_bounds__(128) void mha_proj_gemm(
    const float* __restrict__ A, const float* __restrict__ W,
    const float* __restrict__ bias, _Float16* __restrict__ outH,
    float* __restrict__ outF) {
  __shared__ _Float16 sA[64 * 32];   // [m][k]
  __shared__ _Float16 sB[64 * 32];   // B-transposed: [n][k]

  const int m0 = blockIdx.x * 64;
  const int n0 = blockIdx.y * 64;
  const int tid = threadIdx.x;
  const int wave = tid >> 5;
  const int lane = tid & 31;

  v8f acc[4] = {v8f{}, v8f{}, v8f{}, v8f{}};

  for (int k0 = 0; k0 < DM; k0 += 32) {
    __syncthreads();
    // Stage A tile: 64 rows x 32 cols f32 -> f16 (512 float4, 4 per thread),
    // packed into one 64-bit DS store each.
#pragma unroll
    for (int it = 0; it < 4; ++it) {
      const int i = tid + it * 128;
      const int row = i >> 3;
      const int c4 = (i & 7) * 4;
      const float4 f = *(const float4*)(A + (size_t)(m0 + row) * DM + k0 + c4);
      h4 hv;
      hv.x = (_Float16)f.x; hv.y = (_Float16)f.y;
      hv.z = (_Float16)f.z; hv.w = (_Float16)f.w;
      *(h4*)(sA + row * 32 + c4) = hv;
    }
    // Stage W tile transposed: W[k0..k0+31][n0..n0+63] -> sB[n][k]
#pragma unroll
    for (int it = 0; it < 4; ++it) {
      const int i = tid + it * 128;
      const int kk = i >> 4;
      const int c4 = (i & 15) * 4;
      const float4 f = *(const float4*)(W + (size_t)(k0 + kk) * DM + n0 + c4);
      sB[(c4 + 0) * 32 + kk] = (_Float16)f.x;
      sB[(c4 + 1) * 32 + kk] = (_Float16)f.y;
      sB[(c4 + 2) * 32 + kk] = (_Float16)f.z;
      sB[(c4 + 3) * 32 + kk] = (_Float16)f.w;
    }
    __syncthreads();

    const v16h a = load_a_frag_f16(sA + (wave * 16) * 32, 32);
#pragma unroll
    for (int nt = 0; nt < 4; ++nt) {
      const v16h b = load_b_frag_bt(sB + (nt * 16) * 32, 32);
      acc[nt] = wmma_f16(a, b, acc[nt]);
    }
  }

  // Epilogue: C layout M = v + (lane<16?0:8), N = lane%16.
  // The outH/outF selector is workgroup-uniform: hoist it out of the loops.
  const int mofs = (lane & 16) ? 8 : 0;
  const int nofs = lane & 15;
  if (outH) {
#pragma unroll
    for (int nt = 0; nt < 4; ++nt) {
      const int col = n0 + nt * 16 + nofs;
      const float bv = bias[col];
#pragma unroll
      for (int v = 0; v < 8; ++v) {
        const int row = m0 + wave * 16 + v + mofs;
        outH[(size_t)row * DM + col] = (_Float16)(acc[nt][v] + bv);
      }
    }
  } else {
#pragma unroll
    for (int nt = 0; nt < 4; ++nt) {
      const int col = n0 + nt * 16 + nofs;
      const float bv = bias[col];
#pragma unroll
      for (int v = 0; v < 8; ++v) {
        const int row = m0 + wave * 16 + v + mofs;
        outF[(size_t)row * DM + col] = acc[nt][v] + bv;
      }
    }
  }
}

// ---------------------------------------------------------------------------
// Fused attention: per (b, h, 32-query tile).  256 threads = 8 waves.
// LDS: 32x2048 f32 scores (256KB) + 32x64 f16 Q + 64x64 f16 K/V chunk.
// Q/K staging uses GLOBAL_LOAD_ASYNC_TO_LDS_B128 (pure f16 copies);
// V staging transposes through VGPRs.  Weights hit HBM exactly once.
// ---------------------------------------------------------------------------
#define ATT_SMEM (32 * SQ * 4 + 32 * DK * 2 + 64 * DK * 2)

__global__ __launch_bounds__(256) void mha_attention(
    const _Float16* __restrict__ Qh, const _Float16* __restrict__ Kh,
    const _Float16* __restrict__ Vh, float* __restrict__ weights,
    float* __restrict__ ctx) {
  extern __shared__ char smem[];
  float*    sS  = (float*)smem;                        // [32][2048]
  _Float16* sQ  = (_Float16*)(smem + 32 * SQ * 4);     // [32][64]
  _Float16* sKV = sQ + 32 * DK;                        // [64][64]

  const int q0 = blockIdx.x * 32;
  const int h  = blockIdx.y;
  const int b  = blockIdx.z;
  const int tid  = threadIdx.x;
  const int wave = tid >> 5;
  const int lane = tid & 31;

  const size_t rowBase = (size_t)b * SQ;   // row index base into [B*S, DM]
  const int    colBase = h * DK;           // this head's column slice

  // Stage Q tile (32x64 f16 = 4KB): 256 async b128 copies, 1 per thread.
  {
    const int r = tid >> 3;            // query row 0..31
    const int c8 = (tid & 7) * 8;      // 8 halves = 16 bytes
    async_copy_b128(Qh + (rowBase + q0 + r) * DM + colBase + c8,
                    sQ + r * DK + c8);
  }

  const int qrow = (wave & 1) * 16;   // wave's 16-row slice of the 32-q tile
  const int kcol = (wave >> 1) * 16;  // wave's 16-col slice of the 64-key chunk

  // ---- Phase 1: scores = Q K^T / 8, parked in LDS --------------------------
  for (int c = 0; c < SQ / 64; ++c) {
    // Stage 64-key chunk (8KB): K rows are already B^T layout ([key][feat]).
    // 512 async b128 copies, 2 per thread.
#pragma unroll
    for (int it = 0; it < 2; ++it) {
      const int i = tid + it * 256;
      const int r = i >> 3;            // key row within chunk
      const int c8 = (i & 7) * 8;
      async_copy_b128(Kh + (rowBase + c * 64 + r) * DM + colBase + c8,
                      sKV + r * DK + c8);
    }
    async_wait0();
    __syncthreads();

    v8f acc = v8f{};
#pragma unroll
    for (int kk = 0; kk < DK; kk += 32) {
      const v16h a = load_a_frag_f16(sQ + qrow * DK + kk, DK);
      const v16h bb = load_b_frag_bt(sKV + kcol * DK + kk, DK);
      acc = wmma_f16(a, bb, acc);
    }
    const int mofs = (lane & 16) ? 8 : 0;
    const int nofs = lane & 15;
#pragma unroll
    for (int v = 0; v < 8; ++v) {
      sS[(qrow + v + mofs) * SQ + c * 64 + kcol + nofs] = acc[v] * 0.125f;
    }
    __syncthreads();   // scores written; safe to restage sKV next iter
  }

  // ---- Phase 2: softmax per row; write weights to HBM once -----------------
#pragma unroll
  for (int rr = 0; rr < 4; ++rr) {
    const int row = wave + rr * 8;
    float* Srow = sS + row * SQ;
    float mx = -INFINITY;
    for (int cc = lane; cc < SQ; cc += 32) mx = fmaxf(mx, Srow[cc]);
#pragma unroll
    for (int m = 16; m; m >>= 1) mx = fmaxf(mx, __shfl_xor(mx, m, 32));
    float sum = 0.f;
    for (int cc = lane; cc < SQ; cc += 32) {
      const float e = __expf(Srow[cc] - mx);
      Srow[cc] = e;
      sum += e;
    }
#pragma unroll
    for (int m = 16; m; m >>= 1) sum += __shfl_xor(sum, m, 32);
    const float inv = 1.f / sum;
    float* wrow = weights + ((size_t)(b * NH + h) * SQ + q0 + row) * SQ;
    for (int cc = lane; cc < SQ; cc += 32) {
      const float p = Srow[cc] * inv;
      Srow[cc] = p;
      wrow[cc] = p;
    }
  }
  __syncthreads();

  // ---- Phase 3: context = P V (P from LDS f32 -> f16 fragments) ------------
  const int jcol = kcol;  // wave's 16-col slice of the 64-wide head dim
  v8f acc2 = v8f{};
  for (int c = 0; c < SQ / 64; ++c) {
    // Stage V chunk transposed into sKV: sKV[j][key] = V[key][j]
#pragma unroll
    for (int it = 0; it < 4; ++it) {
      const int i = tid + it * 256;
      const int r = i >> 4;           // key within chunk
      const int c4 = (i & 15) * 4;    // feature j
      const h4 v = *(const h4*)(Vh + (rowBase + c * 64 + r) * DM + colBase + c4);
      sKV[(c4 + 0) * 64 + r] = v.x;
      sKV[(c4 + 1) * 64 + r] = v.y;
      sKV[(c4 + 2) * 64 + r] = v.z;
      sKV[(c4 + 3) * 64 + r] = v.w;
    }
    __syncthreads();
#pragma unroll
    for (int kk = 0; kk < 64; kk += 32) {
      const v16h a = load_a_frag_f32(sS + qrow * SQ + c * 64 + kk, SQ);
      const v16h bb = load_b_frag_bt(sKV + jcol * 64 + kk, 64);
      acc2 = wmma_f16(a, bb, acc2);
    }
    __syncthreads();   // compute done; safe to restage sKV next iter
  }
  // Store context in [B, S, DM] layout (heads interleaved in columns).
  {
    const int mofs = (lane & 16) ? 8 : 0;
    const int nofs = lane & 15;
#pragma unroll
    for (int v = 0; v < 8; ++v) {
      const int row = q0 + qrow + v + mofs;
      const int col = colBase + jcol + nofs;
      ctx[(rowBase + row) * DM + col] = acc2[v];
    }
  }
}

// ---------------------------------------------------------------------------
// Host launcher
// ---------------------------------------------------------------------------
extern "C" void kernel_launch(void* const* d_in, const int* in_sizes, int n_in,
                              void* d_out, int out_size, void* d_ws, size_t ws_size,
                              hipStream_t stream) {
  (void)in_sizes; (void)n_in; (void)out_size; (void)ws_size;

  const float* query = (const float*)d_in[0];
  const float* key   = (const float*)d_in[1];
  const float* value = (const float*)d_in[2];
  const float* Wq = (const float*)d_in[3];
  const float* bq = (const float*)d_in[4];
  const float* Wk = (const float*)d_in[5];
  const float* bk = (const float*)d_in[6];
  const float* Wv = (const float*)d_in[7];
  const float* bv = (const float*)d_in[8];
  const float* Wo = (const float*)d_in[9];
  const float* bo = (const float*)d_in[10];

  float* outF    = (float*)d_out;                          // [B, S, DM]
  float* weights = outF + (size_t)NB * SQ * DM;            // [B, H, S, S]

  const size_t nAct = (size_t)NB * SQ * DM;                // 4096*1024
  _Float16* qh = (_Float16*)d_ws;
  _Float16* kh = qh + nAct;
  _Float16* vh = kh + nAct;
  float*    ctx = (float*)(vh + nAct);                     // total 40 MB

  dim3 gproj(NB * SQ / 64, DM / 64);   // 64 x 16
  mha_proj_gemm<<<gproj, 128, 0, stream>>>(query, Wq, bq, qh, nullptr);
  mha_proj_gemm<<<gproj, 128, 0, stream>>>(key,   Wk, bk, kh, nullptr);
  mha_proj_gemm<<<gproj, 128, 0, stream>>>(value, Wv, bv, vh, nullptr);

  hipFuncSetAttribute((const void*)mha_attention,
                      hipFuncAttributeMaxDynamicSharedMemorySize, ATT_SMEM);
  dim3 gatt(SQ / 32, NH, NB);          // 64 x 16 x 2
  mha_attention<<<gatt, 256, ATT_SMEM, stream>>>(qh, kh, vh, weights, ctx);

  mha_proj_gemm<<<gproj, 128, 0, stream>>>(ctx, Wo, bo, nullptr, outF);
}
